// MechanicsPINN_23837068493026
// MI455X (gfx1250) — compile-verified, matrix operand order checked by reading, not run on hardware
//
#include <hip/hip_runtime.h>
#include <hip/hip_bf16.h>

#define HH   256
#define WW   256
#define BB   64
#define NTOT 65536   // H*W

#define KCHUNK 256                  // K-slice of A staged in LDS
#define KPAD   (KCHUNK + 8)         // +16B row pad: row stride 528B, banks spread

typedef __attribute__((ext_vector_type(16))) __bf16         v16bf;
typedef __attribute__((ext_vector_type(8)))  float          v8f;
typedef __attribute__((ext_vector_type(8)))  unsigned short u16x8;
typedef __attribute__((ext_vector_type(16))) unsigned short u16x16;
typedef __attribute__((ext_vector_type(8)))  unsigned int   u32x8;

// f32 -> bf16 RNE (used only in the tiny layer-3 kernel)
__device__ __forceinline__ unsigned short f2bf(float x) {
    unsigned int u = __builtin_bit_cast(unsigned int, x);
    u += 0x7FFFu + ((u >> 16) & 1u);
    return (unsigned short)(u >> 16);
}

// pack two f32 into one dword of two bf16 (truncation) with a single v_perm_b32
__device__ __forceinline__ unsigned int pack_bf16x2(float f_even, float f_odd) {
    unsigned int lo = __builtin_bit_cast(unsigned int, f_even);
    unsigned int hi = __builtin_bit_cast(unsigned int, f_odd);
    return __builtin_amdgcn_perm(hi, lo, 0x07060302u);
}

// combine two 8x16-bit loads (K..K+7 and K+16..K+23) into one 16-elem vector
#define LD16(p) __builtin_shufflevector(*(const u16x8*)(p), *(const u16x8*)((p) + 16), \
                                        0,1,2,3,4,5,6,7,8,9,10,11,12,13,14,15)

// ---------------- small MLP layers (VALU, negligible cost) ----------------
__global__ void __launch_bounds__(256)
mlp_l1(const float* __restrict__ x, const float* __restrict__ W1,
       const float* __restrict__ b1, float* __restrict__ h1) {
    int idx = blockIdx.x * 256 + threadIdx.x;          // 64*256
    int b = idx >> 8, n = idx & 255;
    float v = x[b * 2 + 0] * W1[n] + x[b * 2 + 1] * W1[256 + n] + b1[n];
    h1[idx] = v > 0.f ? v : 0.f;
}

__global__ void __launch_bounds__(256)
mlp_l2(const float* __restrict__ h1, const float* __restrict__ W2,
       const float* __restrict__ b2, float* __restrict__ h2) {
    int idx = blockIdx.x * 256 + threadIdx.x;          // 64*512
    int b = idx >> 9, n = idx & 511;
    float s = b2[n];
    const float* hr = h1 + b * 256;
#pragma unroll 8
    for (int k = 0; k < 256; ++k) s = fmaf(hr[k], W2[k * 512 + n], s);
    h2[idx] = s > 0.f ? s : 0.f;
}

__global__ void __launch_bounds__(256)
mlp_l3(const float* __restrict__ h2, const float* __restrict__ W3,
       const float* __restrict__ b3, unsigned short* __restrict__ h3bf) {
    int idx = blockIdx.x * 256 + threadIdx.x;          // 64*1024
    int b = idx >> 10, n = idx & 1023;
    float s = b3[n];
    const float* hr = h2 + b * 512;
#pragma unroll 8
    for (int k = 0; k < 512; ++k) s = fmaf(hr[k], W3[k * 1024 + n], s);
    h3bf[idx] = f2bf(s > 0.f ? s : 0.f);               // store bf16 for WMMA A
}

// ---------------- big GEMM: f = h3 @ W4 + b4 via v_wmma_f32_16x16x32_bf16 --
// One wave per 16-wide N tile; full M=64 as 4 accumulators. A (shared by all
// 8 waves of the block) is staged in LDS in 4 K-chunks of 256.
__global__ void __launch_bounds__(256)
gemm4_wmma(const unsigned short* __restrict__ A,   // h3 bf16 [64][1024]
           const float* __restrict__ W4,           // [1024][65536] f32
           const float* __restrict__ b4,           // [65536]
           float* __restrict__ F) {                // [64][65536] f32
    __shared__ unsigned short sA[64 * KPAD];       // ~33 KB

    const int tid   = threadIdx.x;
    const int lane  = tid & 31;
    const int wave  = tid >> 5;
    const int ntile = blockIdx.x * 8 + wave;       // 0..4095
    const int m16   = lane & 15;
    const int khalf = lane >> 4;                   // ISA 16-bit layout half-select
    const int ncol  = ntile * 16 + m16;

    v8f acc0 = {}, acc1 = {}, acc2 = {}, acc3 = {};

    for (int kc = 0; kc < 1024; kc += KCHUNK) {
        if (kc) __syncthreads();                   // previous chunk fully consumed
        // fill sA with A[:, kc:kc+256): 2048 16B units, coalesced, no conflicts
#pragma unroll
        for (int r = 0; r < 8; ++r) {
            int u   = r * 256 + tid;               // 0..2047
            int row = u >> 5;                      // 32 units of 16B per row
            int cu  = u & 31;
            u16x8 v = *(const u16x8*)(A + row * 1024 + kc + cu * 8);
            *(u16x8*)(sA + row * KPAD + cu * 8) = v;
        }
        __syncthreads();

#pragma unroll 2
        for (int k0 = 0; k0 < KCHUNK; k0 += 32) {
            const int kb = k0 + khalf * 8;         // chunk-relative K base
            // B: lane = column n, gather 16 K-rows of W4 (coalesced across lanes)
            const float* bp = W4 + (size_t)(kc + kb) * NTOT + ncol;
            if (kc + k0 + 32 < 1024)               // prefetch next K panel
                __builtin_prefetch(bp + (size_t)32 * NTOT, 0, 0);
            float bf[16];
#pragma unroll
            for (int e = 0; e < 8; ++e) bf[e]     = bp[(size_t)e * NTOT];
#pragma unroll
            for (int e = 0; e < 8; ++e) bf[8 + e] = bp[(size_t)(16 + e) * NTOT];

            // A from LDS: lanes 0-15 = rows M; elems 0-7 -> K=kb.., 8-15 -> K=kb+16..
            const unsigned short* ap = sA + m16 * KPAD + kb;
            v16bf a0 = __builtin_bit_cast(v16bf, (u16x16)LD16(ap));
            v16bf a1 = __builtin_bit_cast(v16bf, (u16x16)LD16(ap + 16 * KPAD));
            v16bf a2 = __builtin_bit_cast(v16bf, (u16x16)LD16(ap + 32 * KPAD));
            v16bf a3 = __builtin_bit_cast(v16bf, (u16x16)LD16(ap + 48 * KPAD));

            // pack 16 f32 -> 8 dwords of bf16 pairs (1 v_perm_b32 per pair)
            u32x8 bpk;
#pragma unroll
            for (int v = 0; v < 8; ++v) bpk[v] = pack_bf16x2(bf[2 * v], bf[2 * v + 1]);
            v16bf bv = __builtin_bit_cast(v16bf, bpk);

            acc0 = __builtin_amdgcn_wmma_f32_16x16x32_bf16(false, a0, false, bv, (short)0, acc0, false, false);
            acc1 = __builtin_amdgcn_wmma_f32_16x16x32_bf16(false, a1, false, bv, (short)0, acc1, false, false);
            acc2 = __builtin_amdgcn_wmma_f32_16x16x32_bf16(false, a2, false, bv, (short)0, acc2, false, false);
            acc3 = __builtin_amdgcn_wmma_f32_16x16x32_bf16(false, a3, false, bv, (short)0, acc3, false, false);
        }
    }

    // C/D layout: VGPR r -> M=r (lanes 0-15) or M=r+8 (lanes 16-31); N = lane&15
    const float bias = b4[ncol];
    const int   mrow = khalf * 8;
#pragma unroll
    for (int r = 0; r < 8; ++r) {
        F[(size_t)( 0 + mrow + r) * NTOT + ncol] = acc0[r] + bias;
        F[(size_t)(16 + mrow + r) * NTOT + ncol] = acc1[r] + bias;
        F[(size_t)(32 + mrow + r) * NTOT + ncol] = acc2[r] + bias;
        F[(size_t)(48 + mrow + r) * NTOT + ncol] = acc3[r] + bias;
    }
}

// ---------------- composed reflect-padded biharmonic residual -------------
__device__ __forceinline__ int refl(int i) {       // jnp 'reflect' (no edge repeat)
    if (i < 0)    i = -i;
    if (i >= 256) i = 510 - i;
    return i;
}
__device__ __forceinline__ float lapAt(const float* __restrict__ f, int y, int x) {
    float c  = f[y * 256 + x];
    float xm = f[y * 256 + refl(x - 1)];
    float xp = f[y * 256 + refl(x + 1)];
    float ym = f[refl(y - 1) * 256 + x];
    float yp = f[refl(y + 1) * 256 + x];
    return (xm + xp - 2.f * c) + (ym + yp - 2.f * c);   // dx = dy = 1
}

__global__ void __launch_bounds__(256)
residual_kernel(const float* __restrict__ F, const float* __restrict__ P,
                float* __restrict__ out) {
    int idx = blockIdx.x * 256 + threadIdx.x;      // B*H*W
    int b = idx >> 16;
    int y = (idx >> 8) & 255;
    int x = idx & 255;
    const float* f = F + (size_t)b * NTOT;

    float lc  = lapAt(f, y, x);
    float lxm = lapAt(f, y, refl(x - 1));
    float lxp = lapAt(f, y, refl(x + 1));
    float lym = lapAt(f, refl(y - 1), x);
    float lyp = lapAt(f, refl(y + 1), x);
    float bih = (lxm + lxp - 2.f * lc) + (lym + lyp - 2.f * lc);

    // EI = GC = KC = 1
    out[idx] = bih + lc + f[y * 256 + x] - P[idx];
}

// ---------------- launcher ------------------------------------------------
extern "C" void kernel_launch(void* const* d_in, const int* in_sizes, int n_in,
                              void* d_out, int out_size, void* d_ws, size_t ws_size,
                              hipStream_t stream) {
    const float* x  = (const float*)d_in[0];
    const float* P  = (const float*)d_in[1];
    const float* W1 = (const float*)d_in[2];
    const float* b1 = (const float*)d_in[3];
    const float* W2 = (const float*)d_in[4];
    const float* b2 = (const float*)d_in[5];
    const float* W3 = (const float*)d_in[6];
    const float* b3 = (const float*)d_in[7];
    const float* W4 = (const float*)d_in[8];
    const float* b4 = (const float*)d_in[9];
    float* out = (float*)d_out;

    char* ws = (char*)d_ws;
    float*          h1 = (float*)(ws);                               //  64 KB
    float*          h2 = (float*)(ws + (64 << 10));                  // 128 KB
    unsigned short* h3 = (unsigned short*)(ws + (192 << 10));        // 128 KB
    float*          F  = (float*)(ws + (320 << 10));                 //  16 MB

    mlp_l1<<<  64, 256, 0, stream>>>(x,  W1, b1, h1);
    mlp_l2<<< 128, 256, 0, stream>>>(h1, W2, b2, h2);
    mlp_l3<<< 256, 256, 0, stream>>>(h2, W3, b3, h3);
    gemm4_wmma<<<512, 256, 0, stream>>>(h3, W4, b4, F);
    residual_kernel<<<(BB * NTOT) / 256, 256, 0, stream>>>(F, P, out);
}
